// AttentionConvolution_25967372271859
// MI455X (gfx1250) — matrix operand order repaired and checked by reading, not compiled, not run on hardware
//
#include <hip/hip_runtime.h>

// CDNA5 / gfx1250: wave32, WMMA 16x16x32 f16 (fp32 accumulate).

typedef __attribute__((ext_vector_type(16))) _Float16 v16h;
typedef __attribute__((ext_vector_type(8)))  float    v8f;

union HFrag { v16h v; _Float16 e[16]; };

#define K_DIM 256   // D_IN
#define D_OUT 64
#define TS 32       // transpose tile

// ------------------------------------------------------------- converts
__global__ void cvt_f32_to_f16(const float* __restrict__ src,
                               _Float16* __restrict__ dst, int n) {
  int i = blockIdx.x * blockDim.x + threadIdx.x;
  if (i < n) dst[i] = (_Float16)src[i];
}

// Transpose-convert X: [K][N] f32 -> [N][K] f16, LDS-tiled so both global
// sides stay coalesced.
__global__ __launch_bounds__(256)
void transpose_cvt(const float* __restrict__ src,   // [K][N]
                   _Float16* __restrict__ dst,      // [N][K]
                   int K, int N) {
  __shared__ _Float16 tile[TS][TS + 1];
  int nt = blockIdx.x * TS;
  int kt = blockIdx.y * TS;
  int tx = threadIdx.x;        // 0..31
  int ty = threadIdx.y;        // 0..7
#pragma unroll
  for (int r = 0; r < TS; r += 8) {
    int k = kt + ty + r;
    int n = nt + tx;
    if (k < K && n < N)
      tile[ty + r][tx] = (_Float16)src[(size_t)k * N + n];
  }
  __syncthreads();
#pragma unroll
  for (int r = 0; r < TS; r += 8) {
    int n = nt + ty + r;
    int k = kt + tx;
    if (n < N && k < K)
      dst[(size_t)n * K + k] = tile[tx][ty + r];
  }
}

// ------------------------------------------------------------- GEMM
// H[n][d] = sum_k W[d][k] * Xt[n][k].  W: [64][256] f16, Xt: [N][256] f16.
// One wave -> 64 rows x 16 cols tile. 8 K-steps x 4 row-tiles = 32 WMMAs.
// Both A and B fragments are 2x contiguous b128 loads per lane.
__global__ __launch_bounds__(256)
void gemm_wmma(const _Float16* __restrict__ Wh,   // [64][256]
               const _Float16* __restrict__ Xt,   // [N][256]
               float* __restrict__ H,             // [N][64]
               int N) {
  int wave = (int)((blockIdx.x * blockDim.x + threadIdx.x) >> 5);
  int lane = threadIdx.x & 31;
  int ngroups = N >> 4;
  if (wave >= ngroups) return;
  int colbase = wave << 4;
  int n   = colbase + (lane & 15);
  int row = lane & 15;
  int hiA = (lane < 16) ? 0 : 16;       // A-frag: K halves {0-15}/{16-31}
  int hiB = (lane < 16) ? 0 : 16;       // B-frag: K halves {0-15}/{16-31}
  const _Float16* xrow = Xt + (size_t)n * K_DIM;

  v8f c0 = {}, c1 = {}, c2 = {}, c3 = {};
  for (int ks = 0; ks < 8; ++ks) {
    HFrag b;
    const _Float16* xk = xrow + ks * 32 + hiB;
#pragma unroll
    for (int j = 0; j < 16; ++j) b.e[j] = xk[j];          // 2x b128
#pragma unroll
    for (int rt = 0; rt < 4; ++rt) {
      HFrag a;
      const _Float16* wr = Wh + (size_t)(rt * 16 + row) * K_DIM + ks * 32 + hiA;
#pragma unroll
      for (int j = 0; j < 16; ++j)
        a.e[j] = wr[j];                                   // 2x b128
      v8f& c = (rt == 0) ? c0 : (rt == 1) ? c1 : (rt == 2) ? c2 : c3;
      c = __builtin_amdgcn_wmma_f32_16x16x32_f16(
          /*neg_a=*/false, a.v, /*neg_b=*/false, b.v,
          /*c_mod=*/(short)0, c, /*reuse_a=*/false, /*reuse_b=*/false);
    }
  }
  // C layout: vgpr v, lanes<16 -> M=v, lanes>=16 -> M=v+8
  int dbase = (lane < 16) ? 0 : 8;
#pragma unroll
  for (int rt = 0; rt < 4; ++rt) {
    v8f c = (rt == 0) ? c0 : (rt == 1) ? c1 : (rt == 2) ? c2 : c3;
#pragma unroll
    for (int v = 0; v < 8; ++v) {
      int d = rt * 16 + dbase + v;
      H[(size_t)n * D_OUT + d] = c[v];
    }
  }
}

// ------------------------------------------------------------- scores
__global__ void score_kernel(const float* __restrict__ H,
                             const float* __restrict__ att0p,
                             const float* __restrict__ att1p,
                             float* __restrict__ a0, float* __restrict__ a1,
                             float* __restrict__ smax, float* __restrict__ den,
                             int N) {
  int nn = blockIdx.x * blockDim.x + threadIdx.x;
  if (nn >= N) return;
  const float* hr = H + (size_t)nn * D_OUT;
  float s0 = 0.f, s1 = 0.f;
#pragma unroll 8
  for (int d = 0; d < D_OUT; ++d) {
    float h = hr[d];
    s0 += h * att0p[d];
    s1 += h * att1p[d];
  }
  a0[nn] = s0;
  a1[nn] = s1;
  smax[nn] = -__builtin_inff();
  den[nn] = 0.f;
}

// ------------------------------------------------------------- edges
__device__ __forceinline__ void atomicMaxF(float* addr, float val) {
  int* ai = (int*)addr;
  int old = __float_as_int(*addr);
  while (__int_as_float(old) < val) {
    int assumed = old;
    old = atomicCAS(ai, assumed, __float_as_int(val));
    if (old == assumed) break;
  }
}

__global__ void edge_pass1(const int* __restrict__ r, const int* __restrict__ c,
                           const float* __restrict__ v,
                           const float* __restrict__ a0,
                           const float* __restrict__ a1,
                           float* __restrict__ s, float* __restrict__ smax,
                           int E) {
  int t = blockIdx.x * blockDim.x + threadIdx.x;
  int stride = gridDim.x * blockDim.x;
  for (; t < E; t += stride) {
    int ri = r[t], ci = c[t];
    float sv = v[t] * (a0[ci] + a1[ri]);
    s[t] = sv;
    atomicMaxF(&smax[ri], sv);
  }
}

__global__ void edge_pass2(const int* __restrict__ r, float* __restrict__ s,
                           const float* __restrict__ smax,
                           float* __restrict__ den, int E) {
  int t = blockIdx.x * blockDim.x + threadIdx.x;
  int stride = gridDim.x * blockDim.x;
  for (; t < E; t += stride) {
    int ri = r[t];
    float e = expf(s[t] - smax[ri]);
    s[t] = e;
    atomicAdd(&den[ri], e);
  }
}

// wave-per-edge: 32 lanes cover the 64 dims -> one contiguous 256B atomic
// burst per edge, L2-resident.
__global__ __launch_bounds__(256)
void edge_pass3(const int* __restrict__ r, const int* __restrict__ c,
                const float* __restrict__ s, const float* __restrict__ den,
                const float* __restrict__ H, float* __restrict__ acc, int E) {
  int wave = (int)((blockIdx.x * blockDim.x + threadIdx.x) >> 5);
  int lane = threadIdx.x & 31;
  int nwaves = (int)((gridDim.x * blockDim.x) >> 5);
  for (int e = wave; e < E; e += nwaves) {
    int ri = r[e], ci = c[e];
    float w = s[e] / den[ri];
    const float* hc = H + (size_t)ci * D_OUT;
    float* ar = acc + (size_t)ri * D_OUT;
    atomicAdd(&ar[lane],      w * hc[lane]);
    atomicAdd(&ar[lane + 32], w * hc[lane + 32]);
  }
}

// ------------------------------------------------------------- epilogue
__global__ void elu_write(const float* __restrict__ acc,
                          float* __restrict__ out, int N) {
  long long i = (long long)blockIdx.x * blockDim.x + threadIdx.x;
  if (i >= (long long)N * D_OUT) return;
  int d = (int)(i / N);
  int nn = (int)(i % N);
  float v = acc[(size_t)nn * D_OUT + d];
  out[i] = (v > 0.f) ? v : (expf(v) - 1.f);
}

// ------------------------------------------------------------- launch
extern "C" void kernel_launch(void* const* d_in, const int* in_sizes, int n_in,
                              void* d_out, int out_size, void* d_ws,
                              size_t ws_size, hipStream_t stream) {
  const float* x    = (const float*)d_in[0];   // [256, N]
  const float* Wm   = (const float*)d_in[1];   // [2,4,64,256]
  const float* att0 = (const float*)d_in[2];   // [5,4,64]
  const float* att1 = (const float*)d_in[3];   // [5,4,64]
  const int*   er   = (const int*)d_in[4];     // [5,E]
  const int*   ec   = (const int*)d_in[5];     // [5,E]
  const float* ev   = (const float*)d_in[6];   // [5,E]

  int N = in_sizes[0] / K_DIM;
  int E = in_sizes[4] / 5;

  char* wsp = (char*)d_ws;
  auto carve = [&](size_t bytes) {
    char* p = wsp;
    wsp += (bytes + 255) & ~(size_t)255;
    return p;
  };
  _Float16* xt   = (_Float16*)carve((size_t)N * K_DIM * sizeof(_Float16));
  _Float16* wh   = (_Float16*)carve((size_t)in_sizes[1] * sizeof(_Float16));
  float*    H    = (float*)carve((size_t)N * D_OUT * sizeof(float));
  float*    acc  = (float*)carve((size_t)N * D_OUT * sizeof(float));
  float*    a0   = (float*)carve((size_t)N * sizeof(float));
  float*    a1   = (float*)carve((size_t)N * sizeof(float));
  float*    smax = (float*)carve((size_t)N * sizeof(float));
  float*    den  = (float*)carve((size_t)N * sizeof(float));
  float*    sbuf = (float*)carve((size_t)E * sizeof(float));

  // X -> node-major f16 (once), W -> f16 (once)
  dim3 tb(TS, 8);
  dim3 tg((N + TS - 1) / TS, (K_DIM + TS - 1) / TS);
  transpose_cvt<<<tg, tb, 0, stream>>>(x, xt, K_DIM, N);
  cvt_f32_to_f16<<<(in_sizes[1] + 255) / 256, 256, 0, stream>>>(Wm, wh,
                                                                in_sizes[1]);

  const int offs[2] = {0, 2};
  const int Pm[2]   = {2, 3};
  int ngroups = N / 16;
  int gemm_blocks = (ngroups + 7) / 8;           // 8 waves per block
  int eb = (E + 255) / 256;
  if (eb > 4096) eb = 4096;

  for (int m = 0; m < 2; ++m) {
    for (int i = 0; i < 4; ++i) {
      const _Float16* Whmi = wh + (size_t)((m * 4 + i) * D_OUT) * K_DIM;
      gemm_wmma<<<gemm_blocks, 256, 0, stream>>>(Whmi, xt, H, N);
      hipMemsetAsync(acc, 0, (size_t)N * D_OUT * sizeof(float), stream);
      for (int k = 0; k < Pm[m]; ++k) {
        int p = offs[m] + k;
        score_kernel<<<(N + 255) / 256, 256, 0, stream>>>(
            H, att0 + (size_t)(p * 4 + i) * D_OUT,
            att1 + (size_t)(p * 4 + i) * D_OUT, a0, a1, smax, den, N);
        const int*   rp = er + (size_t)p * E;
        const int*   cp = ec + (size_t)p * E;
        const float* vp = ev + (size_t)p * E;
        edge_pass1<<<eb, 256, 0, stream>>>(rp, cp, vp, a0, a1, sbuf, smax, E);
        edge_pass2<<<eb, 256, 0, stream>>>(rp, sbuf, smax, den, E);
        edge_pass3<<<4096, 256, 0, stream>>>(rp, cp, sbuf, den, H, acc, E);
      }
      float* outp = (float*)d_out + (size_t)((m * 4 + i) * D_OUT) * N;
      long long tot = (long long)N * D_OUT;
      elu_write<<<(int)((tot + 255) / 256), 256, 0, stream>>>(acc, outp, N);
    }
  }
}
